// LossFunction_6923487282626
// MI455X (gfx1250) — compile-verified, matrix operand order checked by reading, not run on hardware
//
#include <hip/hip_runtime.h>
#include <hip/hip_bf16.h>

typedef __attribute__((ext_vector_type(16))) _Float16 v16h;
typedef __attribute__((ext_vector_type(8)))  _Float16 v8h;
typedef __attribute__((ext_vector_type(4)))  _Float16 v4h;
typedef __attribute__((ext_vector_type(8)))  float    v8f;
typedef __attribute__((ext_vector_type(4)))  float    v4f;

#define B_N 4096
#define P_N 32
#define C_N 512
// 2 / (ALPHA * NORM) = 2/20
#define INV_HALF_SCALE 0.1f

// ---------------- reduction helpers (wave32) ----------------

__device__ __forceinline__ float waveReduceSum(float v) {
#pragma unroll
  for (int off = 16; off > 0; off >>= 1) v += __shfl_xor(v, off, 32);
  return v;
}

// Deterministic block reduction: fixed-order LDS sum by thread 0.
// Result valid in thread 0 only. Safe to call repeatedly (trailing barrier).
__device__ __forceinline__ float blockReduceSum(float v, float* smem, int nwaves) {
  const int lane = threadIdx.x & 31;
  const int wid  = threadIdx.x >> 5;
  v = waveReduceSum(v);
  if (lane == 0) smem[wid] = v;
  __syncthreads();
  float r = 0.f;
  if (threadIdx.x == 0) {
    for (int i = 0; i < nwaves; ++i) r += smem[i];
  }
  __syncthreads();
  return r;
}

// ---------------- WMMA helpers ----------------
// 16-bit A-matrix 16x32 layout (ISA 7.12.2): lane L<16 holds row M=L,
// K = 0..7 and 16..23; lane L+16 holds row M=L, K = 8..15 and 24..31.
// For X·Xᵀ the same per-lane gather serves both the A and B fragment.
__device__ __forceinline__ v16h load_frag(const _Float16* base, int ld) {
  const int lane = threadIdx.x & 31;
  const int r  = lane & 15;
  const int kh = (lane >> 4) << 3;           // 0 or 8
  const _Float16* p = base + r * ld + kh;
  v8h lo = *(const v8h*)(p);                 // K = kh .. kh+7
  v8h hi = *(const v8h*)(p + 16);            // K = 16+kh .. 16+kh+7
  return __builtin_shufflevector(lo, hi, 0,1,2,3,4,5,6,7,8,9,10,11,12,13,14,15);
}

__device__ __forceinline__ v8f wmma16x16x32(v16h a, v16h b, v8f c) {
  // (neg_a, A, neg_b, B, c_mod, C, reuse_a, reuse_b)
  return __builtin_amdgcn_wmma_f32_16x16x32_f16(false, a, false, b, (short)0, c,
                                                false, false);
}

// ---------------- 1) normalize: f -> f16 (normalized), pivots -> inv-norm ---

__global__ void __launch_bounds__(256)
norm_kernel(const float* __restrict__ feats, const float* __restrict__ pfeat,
            _Float16* __restrict__ f_h, float* __restrict__ inv_pf) {
  const int lane = threadIdx.x & 31;
  const int wid  = threadIdx.x >> 5;
  const int row  = blockIdx.x * 8 + wid;       // one wave per row
  const bool isFeat = row < B_N;
  const float* src = isFeat ? (feats + (size_t)row * C_N)
                            : (pfeat + (size_t)(row - B_N) * C_N);
  const int base = lane << 2;                  // coalesced 16B per lane
  v4f x[4];
#pragma unroll
  for (int i = 0; i < 4; ++i) x[i] = *(const v4f*)(src + base + (i << 7));
  float s = 0.f;
#pragma unroll
  for (int i = 0; i < 4; ++i)
#pragma unroll
    for (int j = 0; j < 4; ++j) s = fmaf(x[i][j], x[i][j], s);
  s = waveReduceSum(s);
  const float inv = 1.f / fmaxf(sqrtf(s), 1e-12f);
  if (isFeat) {
    _Float16* dst = f_h + (size_t)row * C_N + base;
#pragma unroll
    for (int i = 0; i < 4; ++i) {
      v4h h;
#pragma unroll
      for (int j = 0; j < 4; ++j) h[j] = (_Float16)(x[i][j] * inv);
      *(v4h*)(dst + (i << 7)) = h;
    }
  } else if (lane == 0) {
    inv_pf[row - B_N] = inv;
  }
}

// ---------------- 2) image-image: f·fᵀ via WMMA, fused L1 loss -------------
// Block tile 64(M) x 256(N), 8 waves = 2x4, each wave 2x4 16x16 WMMA tiles.

__global__ void __launch_bounds__(256)
ii_kernel(const _Float16* __restrict__ f_h, const float* __restrict__ targets,
          float* __restrict__ part) {
  __shared__ _Float16 ldsA[64 * 32];    // 4 KB
  __shared__ _Float16 ldsB[256 * 32];   // 16 KB
  __shared__ float red[8];
  const int t = threadIdx.x;
  const int lane = t & 31;
  const int w = t >> 5;
  const int waveM = w >> 2;             // 0..1
  const int waveN = w & 3;              // 0..3
  const int blockM = blockIdx.y * 64;
  const int blockN = blockIdx.x * 256;

  v8f zero = {};
  v8f acc[2][4];
#pragma unroll
  for (int mi = 0; mi < 2; ++mi)
#pragma unroll
    for (int ni = 0; ni < 4; ++ni) acc[mi][ni] = zero;

  for (int k0 = 0; k0 < C_N; k0 += 32) {
    {   // stage A: 64 rows x 32 halves, one 16B chunk per thread
      const int row = t >> 2;
      const int seg = (t & 3) << 3;
      *(v8h*)&ldsA[row * 32 + seg] =
          *(const v8h*)&f_h[(size_t)(blockM + row) * C_N + k0 + seg];
    }
#pragma unroll
    for (int j = 0; j < 4; ++j) {   // stage B: 256 rows x 32 halves
      const int cid = t + (j << 8);
      const int row = cid >> 2;
      const int seg = (cid & 3) << 3;
      *(v8h*)&ldsB[row * 32 + seg] =
          *(const v8h*)&f_h[(size_t)(blockN + row) * C_N + k0 + seg];
    }
    __syncthreads();
    v16h aF[2], bF[4];
#pragma unroll
    for (int mi = 0; mi < 2; ++mi)
      aF[mi] = load_frag(&ldsA[(waveM * 32 + mi * 16) * 32], 32);
#pragma unroll
    for (int ni = 0; ni < 4; ++ni)
      bF[ni] = load_frag(&ldsB[(waveN * 64 + ni * 16) * 32], 32);
#pragma unroll
    for (int mi = 0; mi < 2; ++mi)
#pragma unroll
      for (int ni = 0; ni < 4; ++ni)
        acc[mi][ni] = wmma16x16x32(aF[mi], bF[ni], acc[mi][ni]);
    __syncthreads();
  }

  // C layout: VGPR v -> M = v + 8*(lane>=16), N = lane&15
  const int nIn  = lane & 15;
  const int mAdd = (lane >> 4) << 3;
  float lsum = 0.f;
#pragma unroll
  for (int mi = 0; mi < 2; ++mi) {
    const int gmBase = blockM + waveM * 32 + mi * 16 + mAdd;
#pragma unroll
    for (int ni = 0; ni < 4; ++ni) {
      const float tn = targets[blockN + waveN * 64 + ni * 16 + nIn];
#pragma unroll
      for (int v = 0; v < 8; ++v) {
        const float tm = targets[gmBase + v];
        const float label = 1.f - INV_HALF_SCALE * fabsf(tm - tn);
        lsum += fabsf(acc[mi][ni][v] - label);
      }
    }
  }
  const float bsum = blockReduceSum(lsum, red, 8);
  if (t == 0) part[blockIdx.y * gridDim.x + blockIdx.x] = bsum;
}

// ---------------- 3) text-text: per-b 32x32 gram via WMMA ------------------
// One block (4 waves) per b. Stage raw f32 tile -> normalized f16 in LDS.

__global__ void __launch_bounds__(128)
tt_kernel(const float* __restrict__ pfeat, const float* __restrict__ inv_pf,
          const float* __restrict__ ps, float* __restrict__ part) {
  __shared__ _Float16 tile[P_N * C_N];  // 32 KB
  __shared__ float red[4];
  const int b = blockIdx.x;
  const int t = threadIdx.x;
  const float* src = pfeat + (size_t)b * P_N * C_N;

#pragma unroll 4
  for (int j = 0; j < 32; ++j) {        // 32*512 halves / (128 thr * 4) = 32
    const int cid = t + (j << 7);
    const int row = cid >> 7;           // 128 4-elem chunks per row
    const int col = (cid & 127) << 2;
    const float s = inv_pf[b * P_N + row];
    v4f x = *(const v4f*)(src + row * C_N + col);
    v4h h;
#pragma unroll
    for (int q = 0; q < 4; ++q) h[q] = (_Float16)(x[q] * s);
    *(v4h*)&tile[row * C_N + col] = h;
  }
  __syncthreads();

  const int w  = t >> 5;
  const int mi = w >> 1, ni = w & 1;    // 2x2 tiles of the 32x32 gram
  v8f acc = {};
  for (int k0 = 0; k0 < C_N; k0 += 32) {
    v16h aF = load_frag(&tile[(mi * 16) * C_N + k0], C_N);
    v16h bF = load_frag(&tile[(ni * 16) * C_N + k0], C_N);
    acc = wmma16x16x32(aF, bF, acc);
  }

  const int lane = t & 31;
  const int nIn  = lane & 15;
  const int mAdd = (lane >> 4) << 3;
  const float pn = ps[b * P_N + ni * 16 + nIn];
  float lsum = 0.f;
#pragma unroll
  for (int v = 0; v < 8; ++v) {
    const float pm = ps[b * P_N + mi * 16 + mAdd + v];
    const float label = 1.f - INV_HALF_SCALE * fabsf(pm - pn);
    lsum += fabsf(acc[v] - label);
  }
  const float bsum = blockReduceSum(lsum, red, 4);
  if (t == 0) part[b] = bsum;
}

// ---------------- 4) image-text: thread-per-(b,p) dot ----------------------

__global__ void __launch_bounds__(256)
it_kernel(const _Float16* __restrict__ f_h, const float* __restrict__ pfeat,
          const float* __restrict__ inv_pf, const float* __restrict__ targets,
          const float* __restrict__ ps, float* __restrict__ part) {
  __shared__ float red[8];
  const int idx = blockIdx.x * 256 + threadIdx.x;   // b*32 + p
  const int b = idx >> 5;
  const _Float16* fr = f_h + (size_t)b * C_N;
  const float* pr = pfeat + (size_t)idx * C_N;
  float dot = 0.f;
  for (int k = 0; k < C_N; k += 8) {
    v8h fh = *(const v8h*)(fr + k);
    v4f x0 = *(const v4f*)(pr + k);
    v4f x1 = *(const v4f*)(pr + k + 4);
#pragma unroll
    for (int q = 0; q < 4; ++q) dot = fmaf((float)fh[q], x0[q], dot);
#pragma unroll
    for (int q = 0; q < 4; ++q) dot = fmaf((float)fh[4 + q], x1[q], dot);
  }
  dot *= inv_pf[idx];
  const float label = 1.f - INV_HALF_SCALE * fabsf(targets[b] - ps[idx]);
  const float bsum = blockReduceSum(fabsf(dot - label), red, 8);
  if (threadIdx.x == 0) part[blockIdx.x] = bsum;
}

// ---------------- 5) mae + final deterministic reduction -------------------

__global__ void __launch_bounds__(256)
mae_kernel(const float* __restrict__ pred, const float* __restrict__ targ,
           float* __restrict__ part) {
  __shared__ float red[8];
  float s = 0.f;
  for (int i = threadIdx.x; i < B_N; i += 256) s += fabsf(pred[i] - targ[i]);
  const float bsum = blockReduceSum(s, red, 8);
  if (threadIdx.x == 0) part[0] = bsum;
}

__global__ void __launch_bounds__(256)
final_kernel(const float* __restrict__ part, float* __restrict__ out) {
  __shared__ float red[8];
  float s_ii = 0.f, s_tt = 0.f, s_it = 0.f;
  for (int i = threadIdx.x; i < 1024; i += 256) s_ii += part[i];
  for (int i = threadIdx.x; i < 4096; i += 256) s_tt += part[1024 + i];
  for (int i = threadIdx.x; i < 512;  i += 256) s_it += part[1024 + 4096 + i];
  s_ii = blockReduceSum(s_ii, red, 8);
  s_tt = blockReduceSum(s_tt, red, 8);
  s_it = blockReduceSum(s_it, red, 8);
  if (threadIdx.x == 0) {
    out[0] = s_ii * (1.f / 16777216.f);   // image_image: mean over B*B
    out[1] = s_tt * (1.f / 4194304.f);    // text_text:   mean over B*P*P
    out[2] = s_it * (1.f / 131072.f);     // image_text:  mean over B*P
    out[3] = part[1024 + 4096 + 512] * (1.f / 4096.f);  // mae
  }
}

// ---------------- launch ----------------------------------------------------

extern "C" void kernel_launch(void* const* d_in, const int* in_sizes, int n_in,
                              void* d_out, int out_size, void* d_ws, size_t ws_size,
                              hipStream_t stream) {
  (void)in_sizes; (void)n_in; (void)out_size; (void)ws_size;
  const float* pred   = (const float*)d_in[0];   // (B,)
  const float* targ   = (const float*)d_in[1];   // (B,)
  const float* feats  = (const float*)d_in[2];   // (B,C)
  const float* pfeat  = (const float*)d_in[3];   // (B,P,C)
  const float* pscore = (const float*)d_in[4];   // (B,P)
  float* out = (float*)d_out;

  // workspace: f16 normalized feats (4 MB) | pivot inv-norms (512 KB) | partials
  _Float16* f_h    = (_Float16*)d_ws;
  float*    inv_pf = (float*)(f_h + (size_t)B_N * C_N);
  float*    part   = inv_pf + (size_t)B_N * P_N;
  // part layout: ii[1024] | tt[4096] | it[512] | mae[1]

  norm_kernel<<<(B_N + B_N * P_N) / 8, 256, 0, stream>>>(feats, pfeat, f_h, inv_pf);
  ii_kernel<<<dim3(B_N / 256, B_N / 64), 256, 0, stream>>>(f_h, targ, part);
  tt_kernel<<<B_N, 128, 0, stream>>>(pfeat, inv_pf, pscore, part + 1024);
  it_kernel<<<(B_N * P_N) / 256, 256, 0, stream>>>(f_h, pfeat, inv_pf, targ, pscore,
                                                   part + 1024 + 4096);
  mae_kernel<<<1, 256, 0, stream>>>(pred, targ, part + 1024 + 4096 + 512);
  final_kernel<<<1, 256, 0, stream>>>(part, out);
}
